// BlockNet_59863254171653
// MI455X (gfx1250) — matrix operand order repaired and checked by reading, not compiled, not run on hardware
//
#include <hip/hip_runtime.h>
#include <hip/hip_bf16.h>

typedef __attribute__((ext_vector_type(16))) _Float16 v16h;
typedef __attribute__((ext_vector_type(8)))  float    v8f;

// ---- VGPR-layout helpers (CDNA5 ISA, 16-bit 16x32 A / 32x16 B) ----
// A: lane<16 holds M=lane, halves map K = {0..7,16..23}; lane>=16 -> K+8
__device__ __forceinline__ int a_klocal(int h, int hi) {
    int v = h >> 1, lo = h & 1;
    int base = (v < 4) ? (2 * v) : (16 + 2 * (v - 4));
    return base + lo + hi * 8;
}
// B: lane<16 -> N=lane, halves map K = {0..7,16..23}; lane>=16 -> N=lane-16, K+8
__device__ __forceinline__ int b_klocal(int h, int hi) { return (h < 8 ? h : h + 8) + hi * 8; }

__device__ __forceinline__ float fast_sigmoid(float z) {
    // 1/(1+e^-z) with v_exp_f32 + v_rcp_f32 (no IEEE divide expansion)
    return __builtin_amdgcn_rcpf(1.f + __expf(-z));
}

// ---- weight packing: combined [Wl(l) | Wp | Wg] -> f16 B fragments ----
// Bf layout: [(l*T + t)*CCH + c] blocks of 512 halves; half index = lane*16 + h
template<int L, int T, int CCH, int LN, int COUT, int X2OFF, int GOFF>
__global__ void prep_kernel(const float* __restrict__ wu,
                            const float* __restrict__ wp,
                            const float* __restrict__ wl,
                            _Float16* __restrict__ Bf) {
    int idx = blockIdx.x * 256 + threadIdx.x;
    constexpr int total = L * T * CCH * 512;
    if (idx >= total) return;
    int h    = idx & 15;
    int lane = (idx >> 4) & 31;
    int rest = idx >> 9;
    int c = rest % CCH; rest /= CCH;
    int t = rest % T;
    int l = rest / T;
    int n = lane & 15, hi = lane >> 4;
    int K   = c * 32 + b_klocal(h, hi);
    int col = t * 16 + n;
    float val = 0.f;
    if (K < LN) {
        if (col < COUT)                              val = wu[(l * LN + K) * COUT + col];
        else if (col >= X2OFF && col < X2OFF + COUT) val = wp[(col - X2OFF) * LN + K];
        else if (col == GOFF)                        val = wl[K];
    }
    Bf[idx] = (_Float16)val;
}

// ---- fused block kernel: gather patches -> WMMA -> gated epilogue ----
template<typename Tin, int CIN, int H, int W, int KK, int S, int OH,
         int COUT, int LN, int CCH, int T, int X2OFF, int GOFF, bool FUSE_FC>
__global__ void __launch_bounds__(256)
block_kernel(const Tin* __restrict__ in, const _Float16* __restrict__ Bf,
             const float* __restrict__ bu, const float* __restrict__ bp,
             const float* __restrict__ bl,
             _Float16* __restrict__ out,
             const float* __restrict__ fcw, const float* __restrict__ fcb,
             float* __restrict__ fcout) {
    constexpr int L = OH * OH;
    __shared__ float lds[8][T][16][17];
    __shared__ float ldsh[(FUSE_FC ? 8 : 1)][16][33];

    const int wid   = threadIdx.x >> 5;
    const int lane  = threadIdx.x & 31;
    const int gwave = blockIdx.x * 8 + wid;
    const int l     = gwave % L;
    const int mt    = gwave / L;
    const int mbase = mt * 16;
    const int i0 = (l / OH) * S, j0 = (l % OH) * S;
    const int n16 = lane & 15, hi = lane >> 4;

    // ---- load A fragments: 16 batch rows x LN features for this location ----
    v16h a[CCH];
    const int b_row = mbase + n16;
#pragma unroll
    for (int c = 0; c < CCH; ++c) {
#pragma unroll
        for (int h = 0; h < 16; ++h) {
            int Kf = c * 32 + a_klocal(h, hi);
            _Float16 v = (_Float16)0.f;
            if (Kf < LN) {
                int ci  = Kf / (KK * KK);
                int rem = Kf % (KK * KK);
                int kh = rem / KK, kw = rem % KK;
                v = (_Float16)(float)in[((size_t)(b_row * CIN + ci) * H + (i0 + kh)) * W + (j0 + kw)];
            }
            a[c][h] = v;
        }
    }

    // ---- WMMA: for each N-tile accumulate over K chunks, stage D to LDS ----
#pragma unroll
    for (int t = 0; t < T; ++t) {
        v8f acc = {};
#pragma unroll
        for (int c = 0; c < CCH; ++c) {
            const v16h bf = *(const v16h*)(Bf + (((size_t)(l * T + t) * CCH + c) << 9) + lane * 16);
            acc = __builtin_amdgcn_wmma_f32_16x16x32_f16(false, a[c], false, bf,
                                                         (short)0, acc, false, false);
        }
#pragma unroll
        for (int r = 0; r < 8; ++r)
            lds[wid][t][r + hi * 8][n16] = acc[r];
    }
    __syncthreads();

    // ---- epilogue: x2 + g*(x1-x2), relu ----
    const float bl0 = bl[0];
    constexpr int NOUT = 16 * COUT;   // multiple of 32 for all blocks
#pragma unroll
    for (int u = 0; u < NOUT / 32; ++u) {
        int idx = u * 32 + lane;
        int m = idx / COUT, co = idx % COUT;
        float x1 = lds[wid][co >> 4][m][co & 15] + bu[co * L + l];
        int c2 = X2OFF + co;
        float x2 = lds[wid][c2 >> 4][m][c2 & 15] + bp[co];
        float g  = fast_sigmoid(lds[wid][GOFF >> 4][m][GOFF & 15] + bl0);
        float y  = fmaf(g, x1 - x2, x2);
        y = y > 0.f ? y : 0.f;
        if constexpr (FUSE_FC) {
            ldsh[wid][m][co] = y;
        } else {
            out[((size_t)(mbase + m) * COUT + co) * L + l] = (_Float16)y;
        }
    }

    if constexpr (FUSE_FC) {   // fused 32->4 FC, writes fp32 d_out
        __syncthreads();
#pragma unroll
        for (int u = 0; u < 2; ++u) {
            int idx = u * 32 + lane;
            int m = idx >> 2, o = idx & 3;
            float s = fcb[o];
#pragma unroll
            for (int cc = 0; cc < 32; ++cc) s = fmaf(ldsh[wid][m][cc], fcw[cc * 4 + o], s);
            fcout[(size_t)(mbase + m) * 4 + o] = s;
        }
    }
}

extern "C" void kernel_launch(void* const* d_in, const int* in_sizes, int n_in,
                              void* d_out, int out_size, void* d_ws, size_t ws_size,
                              hipStream_t stream) {
    (void)in_sizes; (void)n_in; (void)out_size; (void)ws_size;
    const float* x   = (const float*)d_in[0];
    const float* wu1 = (const float*)d_in[1];  const float* bu1 = (const float*)d_in[2];
    const float* wp1 = (const float*)d_in[3];  const float* bp1 = (const float*)d_in[4];
    const float* wl1 = (const float*)d_in[5];  const float* bl1 = (const float*)d_in[6];
    const float* wu2 = (const float*)d_in[7];  const float* bu2 = (const float*)d_in[8];
    const float* wp2 = (const float*)d_in[9];  const float* bp2 = (const float*)d_in[10];
    const float* wl2 = (const float*)d_in[11]; const float* bl2 = (const float*)d_in[12];
    const float* wu3 = (const float*)d_in[13]; const float* bu3 = (const float*)d_in[14];
    const float* wp3 = (const float*)d_in[15]; const float* bp3 = (const float*)d_in[16];
    const float* wl3 = (const float*)d_in[17]; const float* bl3 = (const float*)d_in[18];
    const float* wu4 = (const float*)d_in[19]; const float* bu4 = (const float*)d_in[20];
    const float* wp4 = (const float*)d_in[21]; const float* bp4 = (const float*)d_in[22];
    const float* wl4 = (const float*)d_in[23]; const float* bl4 = (const float*)d_in[24];
    const float* fcw = (const float*)d_in[25]; const float* fcb = (const float*)d_in[26];
    float* out = (float*)d_out;

    // workspace layout (halves)
    _Float16* p   = (_Float16*)d_ws;
    _Float16* Bf1 = p; p += 400 * 1 * 3 * 512;   // 614400
    _Float16* Bf2 = p; p += 81  * 1 * 2 * 512;   // 82944
    _Float16* Bf3 = p; p += 16  * 3 * 2 * 512;   // 49152
    _Float16* Bf4 = p; p += 1   * 5 * 5 * 512;   // 12800
    _Float16* h1  = p; p += (size_t)2048 * 4  * 400;
    _Float16* h2  = p; p += (size_t)2048 * 6  * 81;
    _Float16* h3  = p; p += (size_t)2048 * 16 * 16;

    // weight packing (tiny, L2-resident thereafter)
    prep_kernel<400,1,3,75,4,4,8>    <<<(614400 + 255) / 256, 256, 0, stream>>>(wu1, wp1, wl1, Bf1);
    prep_kernel<81, 1,2,36,6,6,12>   <<<(82944  + 255) / 256, 256, 0, stream>>>(wu2, wp2, wl2, Bf2);
    prep_kernel<16, 3,2,54,16,16,32> <<<(49152  + 255) / 256, 256, 0, stream>>>(wu3, wp3, wl3, Bf3);
    prep_kernel<1,  5,5,144,32,32,64><<<(12800  + 255) / 256, 256, 0, stream>>>(wu4, wp4, wl4, Bf4);

    // fused blocks: grid = (BATCH/16 * L) / 8 waves-per-block
    block_kernel<float,    3,64,64,5,3,20, 4, 75,3,1, 4, 8,false>
        <<<(128 * 400) / 8, 256, 0, stream>>>(x,  Bf1, bu1, bp1, bl1, h1, nullptr, nullptr, nullptr);
    block_kernel<_Float16, 4,20,20,3,2, 9, 6, 36,2,1, 6,12,false>
        <<<(128 * 81)  / 8, 256, 0, stream>>>(h1, Bf2, bu2, bp2, bl2, h2, nullptr, nullptr, nullptr);
    block_kernel<_Float16, 6, 9, 9,3,2, 4,16, 54,2,3,16,32,false>
        <<<(128 * 16)  / 8, 256, 0, stream>>>(h2, Bf3, bu3, bp3, bl3, h3, nullptr, nullptr, nullptr);
    block_kernel<_Float16,16, 4, 4,3,2, 1,32,144,5,5,32,64,true>
        <<<(128 * 1)   / 8, 256, 0, stream>>>(h3, Bf4, bu4, bp4, bl4, nullptr, fcw, fcb, out);
}
